// ProtoDINO_30374008717802
// MI455X (gfx1250) — matrix operand order repaired and checked by reading, not compiled, not run on hardware
//
#include <hip/hip_runtime.h>
#include <math.h>

// ---------------- problem constants ----------------
#define NCLS    200
#define KP      5
#define DIMK    768
#define BSZ     16
#define HW      37
#define NPATCH  (HW*HW)          // 1369
#define NTOT    (BSZ*NPATCH)     // 21904
#define CTOT    ((NCLS+1)*KP)    // 1005
#define CPAD    1024             // padded to 64 n-tiles (divisible by 4-tile blocking)
#define EPS_SK  0.05f
#define TINY    1e-16f
#define GAMMA_  0.99f

typedef __bf16 bf16_t;
typedef __attribute__((ext_vector_type(8)))  bf16_t bf8v;
typedef __attribute__((ext_vector_type(16))) bf16_t bf16v;
typedef __attribute__((ext_vector_type(8)))  float  f8v;

// =====================================================================
// Normalization / precision-conversion kernels
// =====================================================================

// one block per token: l2-normalize 768-dim row -> f32 ptn + bf16 copy (WMMA A operand)
__global__ void k_norm_tokens(const float* __restrict__ X, float* __restrict__ ptn,
                              bf16_t* __restrict__ A1) {
  __shared__ float red[256];
  int i = blockIdx.x;
  int t = threadIdx.x;
  const float* row = X + (size_t)i * DIMK;
  float s = 0.f;
  #pragma unroll
  for (int j = 0; j < 3; ++j) { float x = row[t + 256*j]; s += x*x; }
  red[t] = s; __syncthreads();
  for (int o = 128; o; o >>= 1) { if (t < o) red[t] += red[t+o]; __syncthreads(); }
  float inv = 1.0f / (sqrtf(red[0]) + 1e-12f);
  #pragma unroll
  for (int j = 0; j < 3; ++j) {
    int d = t + 256*j;
    float v = row[d] * inv;
    ptn[(size_t)i*DIMK + d] = v;
    A1[(size_t)i*DIMK + d]  = (bf16_t)v;
  }
}

// one block per (class,proto) row (padded to 1024): l2-normalize -> bf16, stored
// row-major [col][k] which IS the K-contiguous "Bt" operand for the logits GEMM.
__global__ void k_norm_protos(const float* __restrict__ P, bf16_t* __restrict__ Bt2) {
  __shared__ float red[256];
  int rowi = blockIdx.x;            // 0..1023
  int t = threadIdx.x;
  if (rowi >= CTOT) {
    #pragma unroll
    for (int j = 0; j < 3; ++j) Bt2[(size_t)rowi*DIMK + t + 256*j] = (bf16_t)0.0f;
    return;
  }
  const float* row = P + (size_t)rowi * DIMK;
  float s = 0.f;
  #pragma unroll
  for (int j = 0; j < 3; ++j) { float x = row[t + 256*j]; s += x*x; }
  red[t] = s; __syncthreads();
  for (int o = 128; o; o >>= 1) { if (t < o) red[t] += red[t+o]; __syncthreads(); }
  float inv = 1.0f / (sqrtf(red[0]) + 1e-12f);
  #pragma unroll
  for (int j = 0; j < 3; ++j) {
    int d = t + 256*j;
    Bt2[(size_t)rowi*DIMK + d] = (bf16_t)(row[d] * inv);
  }
}

// transpose W into bf16 Bt layout: Wt[j][d] = W[d][j]
__global__ void k_prep_W(const float* __restrict__ W, bf16_t* __restrict__ Wt) {
  int tgt = blockIdx.x*256 + threadIdx.x;
  if (tgt >= DIMK*DIMK) return;
  int d = tgt % DIMK, j = tgt / DIMK;
  Wt[(size_t)j*DIMK + d] = (bf16_t)W[(size_t)d*DIMK + j];
}

// =====================================================================
// BF16 WMMA GEMM, 1x4 register-blocked: one wave owns a 16x64 strip of C.
//   A  : row-major bf16 [M][K]   -> one A fragment reused by 4 WMMAs/k-step
//   Bt : bf16 [Npad][K] (B^T, K-contiguous -> single 16-half load per lane)
// Loads per 4 WMMAs: 2 (A) + 8 (B) b128/lane instead of 16 -> load-port
// pressure below the WMMA issue rate; 4 independent accumulators hide the
// WMMA->WMMA RAW hazard.
// Epilogues:  outBf != 0 : C += residual + bias, store bf16 (adapter GEMM)
//             else       : store f32 to outF (ld=ldOutF), cols < ncols
// =====================================================================
__global__ void k_gemm_wmma(const bf16_t* __restrict__ A, const bf16_t* __restrict__ Bt,
                            int Mtiles, int Ntiles4, int Kdim,
                            const float* __restrict__ residual, const float* __restrict__ bias,
                            bf16_t* __restrict__ outBf,
                            float* __restrict__ outF, int ldOutF, int ncols) {
  int strip = blockIdx.x * blockDim.y + threadIdx.y;   // uniform per wave
  if (strip >= Mtiles * Ntiles4) return;               // whole wave exits -> EXEC all-1
  int mt  = strip / Ntiles4;
  int ntb = (strip % Ntiles4) * 4;                     // first of 4 n-tiles
  int lane = threadIdx.x;
  int half = lane >> 4, l16 = lane & 15;

  const bf16_t* Arow = A + (size_t)(mt*16 + l16) * Kdim;             // A: M=l16
  const bf16_t* B0 = Bt + (size_t)((ntb+0)*16 + l16) * Kdim + half*16; // B: N=l16, Kbase=half*16
  const bf16_t* B1 = Bt + (size_t)((ntb+1)*16 + l16) * Kdim + half*16;
  const bf16_t* B2 = Bt + (size_t)((ntb+2)*16 + l16) * Kdim + half*16;
  const bf16_t* B3 = Bt + (size_t)((ntb+3)*16 + l16) * Kdim + half*16;

  f8v acc0 = {}, acc1 = {}, acc2 = {}, acc3 = {};
  for (int k0 = 0; k0 < Kdim; k0 += 32) {
    bf8v  alo = *(const bf8v*)(Arow + k0 + half*16);       // K = k0 + half*16 + e
    bf8v  ahi = *(const bf8v*)(Arow + k0 + half*16 + 8);   // K = k0 + half*16 + 8 + e
    bf16v av;
    #pragma unroll
    for (int e = 0; e < 8; ++e) { av[e] = alo[e]; av[8+e] = ahi[e]; }
    bf16v bv0 = *(const bf16v*)(B0 + k0);                  // K = k0 + half*16 + e
    bf16v bv1 = *(const bf16v*)(B1 + k0);
    bf16v bv2 = *(const bf16v*)(B2 + k0);
    bf16v bv3 = *(const bf16v*)(B3 + k0);
    acc0 = __builtin_amdgcn_wmma_f32_16x16x32_bf16(false, av, false, bv0, (short)0, acc0, false, false);
    acc1 = __builtin_amdgcn_wmma_f32_16x16x32_bf16(false, av, false, bv1, (short)0, acc1, false, false);
    acc2 = __builtin_amdgcn_wmma_f32_16x16x32_bf16(false, av, false, bv2, (short)0, acc2, false, false);
    acc3 = __builtin_amdgcn_wmma_f32_16x16x32_bf16(false, av, false, bv3, (short)0, acc3, false, false);
  }

  int rbase = mt*16 + half*8;     // C/D: VGPR v -> M = half*8 + v, N = l16
  f8v accs[4] = {acc0, acc1, acc2, acc3};
  if (outBf) {
    #pragma unroll
    for (int j = 0; j < 4; ++j) {
      int col = (ntb + j)*16 + l16;
      #pragma unroll
      for (int v = 0; v < 8; ++v) {
        int row = rbase + v;
        float val = accs[j][v] + residual[(size_t)row*Kdim + col] + bias[col];
        outBf[(size_t)row*Kdim + col] = (bf16_t)val;
      }
    }
  } else {
    #pragma unroll
    for (int j = 0; j < 4; ++j) {
      int col = (ntb + j)*16 + l16;
      if (col < ncols) {
        #pragma unroll
        for (int v = 0; v < 8; ++v)
          outF[(size_t)(rbase + v)*ldOutF + col] = accs[j][v];
      }
    }
  }
}

// =====================================================================
// PCA power iteration (rank-1, centered)
// =====================================================================
__global__ void k_colmean(const float* __restrict__ X, float* __restrict__ mean) {
  int d = blockIdx.x*256 + threadIdx.x; if (d >= DIMK) return;
  float s = 0.f;
  for (int i = 0; i < NTOT; ++i) s += X[(size_t)i*DIMK + d];
  mean[d] = s * (1.0f/(float)NTOT);
}

__global__ void k_init_v(float* __restrict__ v) {
  int d = blockIdx.x*256 + threadIdx.x; if (d >= DIMK) return;
  v[d] = rsqrtf((float)DIMK);
}

// u = (X - mean) @ v : one wave per row
__global__ void k_matvec_u(const float* __restrict__ X, const float* __restrict__ mean,
                           const float* __restrict__ v, float* __restrict__ u) {
  int i = blockIdx.x * blockDim.y + threadIdx.y; if (i >= NTOT) return;
  int lane = threadIdx.x;
  const float* row = X + (size_t)i*DIMK;
  float acc = 0.f;
  for (int d = lane; d < DIMK; d += 32) acc += (row[d] - mean[d]) * v[d];
  for (int off = 16; off; off >>= 1) acc += __shfl_down(acc, off, 32);
  if (lane == 0) u[i] = acc;
}

// v = (X - mean)^T @ u  (folds sum(u) into same loop)
__global__ void k_matvec_v(const float* __restrict__ X, const float* __restrict__ mean,
                           const float* __restrict__ u, float* __restrict__ v) {
  int d = blockIdx.x*256 + threadIdx.x; if (d >= DIMK) return;
  float s = 0.f, su = 0.f;
  for (int i = 0; i < NTOT; ++i) { float ui = u[i]; s += X[(size_t)i*DIMK + d]*ui; su += ui; }
  v[d] = s - mean[d]*su;
}

__global__ void k_norm_v(float* __restrict__ v) {
  __shared__ float red[256];
  int t = threadIdx.x;
  float s = 0.f;
  for (int d = t; d < DIMK; d += 256) { float x = v[d]; s += x*x; }
  red[t] = s; __syncthreads();
  for (int o = 128; o; o >>= 1) { if (t < o) red[t] += red[t+o]; __syncthreads(); }
  float inv = 1.0f / (sqrtf(red[0]) + 1e-12f);
  for (int d = t; d < DIMK; d += 256) v[d] *= inv;
}

__global__ void k_minmax(const float* __restrict__ u, float* __restrict__ stats) {
  __shared__ float smin[256], smax[256];
  int t = threadIdx.x;
  float mn = 3e38f, mx = -3e38f;
  for (int i = t; i < NTOT; i += 256) { float x = u[i]; mn = fminf(mn,x); mx = fmaxf(mx,x); }
  smin[t] = mn; smax[t] = mx; __syncthreads();
  for (int o = 128; o; o >>= 1) {
    if (t < o) { smin[t] = fminf(smin[t], smin[t+o]); smax[t] = fmaxf(smax[t], smax[t+o]); }
    __syncthreads();
  }
  if (t == 0) { stats[0] = smin[0]; stats[1] = smax[0]; }
}

// =====================================================================
// Pseudo labels + factored Sinkhorn (masks are disjoint across classes,
// so row scaler r is per-token, col scaler ccol is per (class,proto))
// =====================================================================
__global__ void k_zero_small(int* __restrict__ cnt, float* __restrict__ S,
                             float* __restrict__ ccol) {
  int t = blockIdx.x*256 + threadIdx.x;
  if (t < 256)  { cnt[t] = 0; S[t] = 0.f; }
  if (t < 1024) ccol[t] = 1.0f;
}

__global__ void k_pseudo(const float* __restrict__ u, const float* __restrict__ stats,
                         const int* __restrict__ labels, int* __restrict__ pl,
                         float* __restrict__ pseudo_out, int* __restrict__ cnt) {
  int i = blockIdx.x*256 + threadIdx.x; if (i >= NTOT) return;
  float us = (u[i] - stats[0]) / (stats[1] - stats[0]);
  int lab = (us <= 0.5f) ? labels[i / NPATCH] : NCLS;
  pl[i] = lab;
  pseudo_out[i] = (float)lab;
  if (lab < NCLS) atomicAdd(&cnt[lab], 1);
}

__global__ void k_rank(const int* __restrict__ cnt, int* __restrict__ rank) {
  if (threadIdx.x == 0 && blockIdx.x == 0) {
    int run = 0;
    for (int c = 0; c < NCLS; ++c) {
      if (cnt[c] > 0) { rank[c] = run; ++run; } else rank[c] = 0;
    }
  }
}

__global__ void k_skS(const float* __restrict__ logits, const int* __restrict__ pl,
                      float* __restrict__ S) {
  int i = blockIdx.x*256 + threadIdx.x; if (i >= NTOT) return;
  int c = pl[i]; if (c >= NCLS) return;
  const float* lp = logits + (size_t)i*CTOT + c*KP;
  float s = 0.f;
  #pragma unroll
  for (int k = 0; k < KP; ++k) s += expf(lp[k] * (1.0f/EPS_SK));
  atomicAdd(&S[c], s);
}

__global__ void k_skInitR(const int* __restrict__ pl, const float* __restrict__ S,
                          float* __restrict__ r) {
  int i = blockIdx.x*256 + threadIdx.x; if (i >= NTOT) return;
  int c = pl[i];
  r[i] = (c < NCLS) ? 1.0f / (S[c] + TINY) : 0.0f;
}

__global__ void k_zero_colsum(float* __restrict__ cs) {
  int t = blockIdx.x*256 + threadIdx.x; if (t < 1024) cs[t] = 0.f;
}

__global__ void k_skCol(const float* __restrict__ logits, const int* __restrict__ pl,
                        const float* __restrict__ r, float* __restrict__ cs) {
  int i = blockIdx.x*256 + threadIdx.x; if (i >= NTOT) return;
  int c = pl[i]; if (c >= NCLS) return;
  float ri = r[i];
  const float* lp = logits + (size_t)i*CTOT + c*KP;
  #pragma unroll
  for (int k = 0; k < KP; ++k)
    atomicAdd(&cs[c*KP + k], expf(lp[k] * (1.0f/EPS_SK)) * ri);
}

__global__ void k_skColUp(float* __restrict__ ccol, const float* __restrict__ cs) {
  int t = blockIdx.x*256 + threadIdx.x; if (t >= NCLS*KP) return;
  float s = ccol[t] * cs[t];                       // actual column sum of current L
  ccol[t] = ccol[t] / (s + TINY) * (1.0f/(float)KP);
}

__global__ void k_skRow(const float* __restrict__ logits, const int* __restrict__ pl,
                        const float* __restrict__ ccol, const int* __restrict__ cnt,
                        float* __restrict__ r) {
  int i = blockIdx.x*256 + threadIdx.x; if (i >= NTOT) return;
  int c = pl[i]; if (c >= NCLS) return;
  const float* lp = logits + (size_t)i*CTOT + c*KP;
  float ri = r[i], s = 0.f;
  #pragma unroll
  for (int k = 0; k < KP; ++k) s += expf(lp[k] * (1.0f/EPS_SK)) * ccol[c*KP + k];
  s *= ri;
  float Ncc = fmaxf((float)cnt[c], 1.0f);
  r[i] = ri / (s + TINY) / Ncc;
}

// final assignment + materialize per-token weights Aw[i][k] = Nc * L0 * r * c
__global__ void k_assign(const float* __restrict__ logits, const int* __restrict__ pl,
                         const float* __restrict__ ccol, const float* __restrict__ r,
                         const int* __restrict__ cnt, const int* __restrict__ rank,
                         float* __restrict__ assign_out, float* __restrict__ Aw) {
  int i = blockIdx.x*256 + threadIdx.x; if (i >= NTOT) return;
  int c = pl[i];
  if (c >= NCLS) {
    assign_out[i] = -1.0f;
    #pragma unroll
    for (int k = 0; k < KP; ++k) Aw[(size_t)i*KP + k] = 0.f;
    return;
  }
  float Ncc = fmaxf((float)cnt[c], 1.0f);
  float ri = r[i];
  const float* lp = logits + (size_t)i*CTOT + c*KP;
  float best = -1.0f; int bk = 0;
  #pragma unroll
  for (int k = 0; k < KP; ++k) {
    float a = expf(lp[k] * (1.0f/EPS_SK)) * ri * ccol[c*KP + k] * Ncc;
    Aw[(size_t)i*KP + k] = a;
    if (a > best) { best = a; bk = k; }              // first max wins (jnp.argmax)
  }
  assign_out[i] = (float)(bk + KP * rank[c]);
}

// =====================================================================
// Prototype EMA update
// =====================================================================
__global__ void k_zeroPacc(float* __restrict__ Pacc) {
  int t = blockIdx.x*256 + threadIdx.x;
  if (t < NCLS*KP*DIMK) Pacc[t] = 0.f;
}

// grid (200, 3): block = one class x 256 dims; skip absent classes
__global__ void k_pnew(const int* __restrict__ pl, const int* __restrict__ cnt,
                       const float* __restrict__ Aw, const float* __restrict__ ptn,
                       float* __restrict__ Pacc) {
  int c = blockIdx.x;
  if (cnt[c] == 0) return;
  int d = blockIdx.y*256 + threadIdx.x;
  float a0=0.f,a1=0.f,a2=0.f,a3=0.f,a4=0.f;
  for (int i = 0; i < NTOT; ++i) {
    if (pl[i] != c) continue;                       // scalar broadcast check
    float x = ptn[(size_t)i*DIMK + d];
    const float* w = Aw + (size_t)i*KP;
    a0 += w[0]*x; a1 += w[1]*x; a2 += w[2]*x; a3 += w[3]*x; a4 += w[4]*x;
  }
  size_t base = ((size_t)c*KP)*DIMK + d;
  Pacc[base            ] = a0;
  Pacc[base +   DIMK   ] = a1;
  Pacc[base + 2*DIMK   ] = a2;
  Pacc[base + 3*DIMK   ] = a3;
  Pacc[base + 4*DIMK   ] = a4;
}

__global__ void k_ema(const float* __restrict__ protos, const float* __restrict__ Pacc,
                      const int* __restrict__ cnt, float* __restrict__ outP) {
  int t = blockIdx.x*256 + threadIdx.x;
  if (t >= (NCLS+1)*KP*DIMK) return;
  int c = t / (KP*DIMK);
  float p = protos[t];
  if (c < NCLS && cnt[c] > 0) p = GAMMA_*p + (1.0f-GAMMA_)*Pacc[t];
  outP[t] = p;
}

// =====================================================================
// Pooling + sa classification head
// =====================================================================
__global__ void k_pooled(const float* __restrict__ logits, float* __restrict__ pooled) {
  int t = blockIdx.x*256 + threadIdx.x; if (t >= BSZ*CTOT) return;
  int b = t / CTOT, ck = t % CTOT;
  const float* base = logits + (size_t)b*NPATCH*CTOT + ck;
  float s = 0.f;
  for (int n = 0; n < NPATCH; ++n) s += base[(size_t)n*CTOT];
  pooled[t] = s * (1.0f/(float)NPATCH);
}

__global__ void k_head(const float* __restrict__ pooled, const float* __restrict__ sa,
                       float* __restrict__ out) {
  int t = blockIdx.x*256 + threadIdx.x; if (t >= BSZ*NCLS) return;
  int b = t / NCLS, c = t % NCLS;
  float m = -3e38f;
  #pragma unroll
  for (int k = 0; k < KP; ++k) m = fmaxf(m, sa[c*KP + k]);
  float e[KP], s = 0.f;
  #pragma unroll
  for (int k = 0; k < KP; ++k) { e[k] = expf(sa[c*KP + k] - m); s += e[k]; }
  float acc = 0.f;
  #pragma unroll
  for (int k = 0; k < KP; ++k)
    acc += pooled[(size_t)b*CTOT + c*KP + k] * (e[k]/s * (float)KP);
  out[(size_t)b*NCLS + c] = acc;
}

// =====================================================================
// Host-side orchestration
// =====================================================================
extern "C" void kernel_launch(void* const* d_in, const int* in_sizes, int n_in,
                              void* d_out, int out_size, void* d_ws, size_t ws_size,
                              hipStream_t stream) {
  (void)in_sizes; (void)n_in; (void)out_size; (void)ws_size;
  const float* X      = (const float*)d_in[0];   // [16,1369,768]
  const int*   labels = (const int*)  d_in[1];   // [16]
  const float* P      = (const float*)d_in[2];   // [201,5,768]
  const float* W      = (const float*)d_in[3];   // [768,768]
  const float* bvec   = (const float*)d_in[4];   // [768]
  const float* sa     = (const float*)d_in[5];   // [200,5]

  // output layout (flat f32, return order)
  float* out = (float*)d_out;
  float* out_cls    = out;                                  // 3200
  float* out_logits = out + 3200;                           // 22,013,520
  float* out_pooled = out_logits + (size_t)NTOT*CTOT;       // 16,080
  float* out_assign = out_pooled + BSZ*CTOT;                // 21,904
  float* out_proto  = out_assign + NTOT;                    // 771,840
  float* out_pseudo = out_proto + (size_t)(NCLS+1)*KP*DIMK; // 21,904

  // workspace carving (256B aligned); total ~142 MB
  char* ws = (char*)d_ws;
  size_t off = 0;
  auto carve = [&](size_t bytes) -> char* {
    char* p = ws + off;
    off = (off + bytes + 255) & ~(size_t)255;
    return p;
  };
  float*  ptn   = (float*) carve((size_t)NTOT*DIMK*4);        // normalized tokens f32
  bf16_t* A1    = (bf16_t*)carve((size_t)NTOT*DIMK*2);        // ptn bf16 (GEMM1 A)
  bf16_t* A2    = (bf16_t*)carve((size_t)NTOT*DIMK*2);        // updated bf16 (GEMM2 A)
  bf16_t* Wt    = (bf16_t*)carve((size_t)DIMK*DIMK*2);        // W^T bf16 (GEMM1 Bt)
  bf16_t* Bt2   = (bf16_t*)carve((size_t)CPAD*DIMK*2);        // protn bf16 (GEMM2 Bt)
  float*  u     = (float*) carve((size_t)NTOT*4);
  float*  v     = (float*) carve((size_t)DIMK*4);
  float*  mean  = (float*) carve((size_t)DIMK*4);
  float*  stats = (float*) carve(64*4);
  int*    pl    = (int*)   carve((size_t)NTOT*4);
  int*    cnt   = (int*)   carve(256*4);
  int*    rankb = (int*)   carve(256*4);
  float*  Ssum  = (float*) carve(256*4);
  float*  r     = (float*) carve((size_t)NTOT*4);
  float*  ccol  = (float*) carve(1024*4);
  float*  csum  = (float*) carve(1024*4);
  float*  Aw    = (float*) carve((size_t)NTOT*KP*4);
  float*  Pacc  = (float*) carve((size_t)NCLS*KP*DIMK*4);

  const int TB = 256;
  const int gTok = (NTOT + TB - 1) / TB;     // 86

  // ---- normalization + bf16 prep ----
  k_norm_tokens<<<NTOT, TB, 0, stream>>>(X, ptn, A1);
  k_norm_protos<<<CPAD, TB, 0, stream>>>(P, Bt2);
  k_prep_W<<<(DIMK*DIMK + TB - 1)/TB, TB, 0, stream>>>(W, Wt);

  // ---- GEMM1: updated = ptn + ptn@W + b  (bf16 WMMA, 16x64 strips, fused epilogue) ----
  {
    int mt = NTOT/16, nt4 = (DIMK/16)/4;            // 1369 x 12 strips
    int strips = mt*nt4;
    dim3 blk(32, 8);
    k_gemm_wmma<<<(strips + 7)/8, blk, 0, stream>>>(A1, Wt, mt, nt4, DIMK,
                                                    ptn, bvec, A2,
                                                    nullptr, 0, 0);
  }
  // ---- GEMM2: logits = updated @ protn^T  (bf16 WMMA -> f32 out) ----
  {
    int mt = NTOT/16, nt4 = (CPAD/16)/4;            // 1369 x 16 strips
    int strips = mt*nt4;
    dim3 blk(32, 8);
    k_gemm_wmma<<<(strips + 7)/8, blk, 0, stream>>>(A2, Bt2, mt, nt4, DIMK,
                                                    nullptr, nullptr, nullptr,
                                                    out_logits, CTOT, CTOT);
  }

  // ---- PCA rank-1 power iteration (10 iters, exact reference recipe) ----
  k_colmean<<<3, TB, 0, stream>>>(X, mean);
  k_init_v<<<3, TB, 0, stream>>>(v);
  {
    dim3 blk(32, 8);
    int gRow = (NTOT + 7)/8;
    for (int it = 0; it < 10; ++it) {
      k_matvec_u<<<gRow, blk, 0, stream>>>(X, mean, v, u);
      k_matvec_v<<<3, TB, 0, stream>>>(X, mean, u, v);
      k_norm_v<<<1, TB, 0, stream>>>(v);
    }
    k_matvec_u<<<gRow, blk, 0, stream>>>(X, mean, v, u);
    // final u normalization is a positive scaling -> min-max scaling invariant, skip
  }
  k_minmax<<<1, TB, 0, stream>>>(u, stats);

  // ---- pseudo labels ----
  k_zero_small<<<4, TB, 0, stream>>>(cnt, Ssum, ccol);
  k_pseudo<<<gTok, TB, 0, stream>>>(u, stats, labels, pl, out_pseudo, cnt);
  k_rank<<<1, 32, 0, stream>>>(cnt, rankb);

  // ---- factored Sinkhorn: L = diag(r) * exp(logit/eps) * diag(c) ----
  k_skS<<<gTok, TB, 0, stream>>>(out_logits, pl, Ssum);
  k_skInitR<<<gTok, TB, 0, stream>>>(pl, Ssum, r);
  for (int it = 0; it < 3; ++it) {
    k_zero_colsum<<<4, TB, 0, stream>>>(csum);
    k_skCol<<<gTok, TB, 0, stream>>>(out_logits, pl, r, csum);
    k_skColUp<<<4, TB, 0, stream>>>(ccol, csum);
    k_skRow<<<gTok, TB, 0, stream>>>(out_logits, pl, ccol, cnt, r);
  }
  k_assign<<<gTok, TB, 0, stream>>>(out_logits, pl, ccol, r, cnt, rankb,
                                    out_assign, Aw);

  // ---- prototype EMA ----
  k_zeroPacc<<<(NCLS*KP*DIMK + TB - 1)/TB, TB, 0, stream>>>(Pacc);
  {
    dim3 grid(NCLS, DIMK/TB);   // 200 x 3
    k_pnew<<<grid, TB, 0, stream>>>(pl, cnt, Aw, ptn, Pacc);
  }
  k_ema<<<((NCLS+1)*KP*DIMK + TB - 1)/TB, TB, 0, stream>>>(P, Pacc, cnt, out_proto);

  // ---- pooling + head ----
  k_pooled<<<(BSZ*CTOT + TB - 1)/TB, TB, 0, stream>>>(out_logits, out_pooled);
  k_head<<<(BSZ*NCLS + TB - 1)/TB, TB, 0, stream>>>(out_pooled, sa, out_cls);
}
